// MapNet_yolo_25409026523958
// MI455X (gfx1250) — compile-verified, matrix operand order checked by reading, not run on hardware
//
#include <hip/hip_runtime.h>
#include <hip/hip_bf16.h>
#include <math.h>

// ---------------------------------------------------------------------------
// MapNet scan on MI455X (gfx1250). All three GEMM-shaped stages run on
// v_wmma_f32_16x16x32_bf16 (bf16 operands, f32 accumulate):
//   conv1 (write-memory): M=32ch x N=15625px x K=8ang*625tap
//   conv2 (localize):     M=8ang(pad16) x N=15625px x K=31ch*625tap
//   GRU gates:            M=96gate x N=125000px x K=32ch (single chunk)
// ~4e11 FLOP vs ~80 MB working set (fits 192 MB L2) -> matrix-core bound.
// K regrouped in (chan,ky) blocks of 32 (25 taps + 7 ZERO pad, so the B-side
// needs no masking at all); N-tiles are 16 consecutive pixels of one row so
// B per chunk is one contiguous LDS-staged row; each wave computes TWO
// N-tiles per A fragment (halves A-side L2 traffic).
// ---------------------------------------------------------------------------

#define LSEQ 6
#define BS   8
#define FCH  32
#define MAPD 125
#define LOCD 25
#define NANG 8
#define PADR 12
#define NPIX (MAPD * MAPD)              // 15625
#define KG   32                          // padded kx per (chan,ky) group
#define KP1  (NANG * LOCD * KG)          // 6400  conv1 A-row stride
#define KP2  ((FCH - 1) * LOCD * KG)     // 24800 conv2 A-row stride
#define LROW 160                         // padded LDS row (x+kx <= 127+31=158)

typedef __attribute__((ext_vector_type(16))) __bf16 v16bf;
typedef __attribute__((ext_vector_type(8)))  float  v8f;

static __device__ __forceinline__ v8f wmma_bf16(v16bf a, v16bf b, v8f c) {
  return __builtin_amdgcn_wmma_f32_16x16x32_bf16(false, a, false, b, (short)0,
                                                 c, false, false);
}

// ---------------------------------------------------------------------------
__global__ void __launch_bounds__(256)
zero_bf16_kernel(__bf16* __restrict__ p, long n) {
  long i = (long)blockIdx.x * blockDim.x + threadIdx.x;
  if (i < n) p[i] = (__bf16)0.0f;
}

__global__ void __launch_bounds__(256)
copy_f32_kernel(const float* __restrict__ src, float* __restrict__ dst, long n) {
  long i = (long)blockIdx.x * blockDim.x + threadIdx.x;
  if (i < n) dst[i] = src[i];
}

__global__ void __launch_bounds__(256)
init_pose_kernel(__bf16* __restrict__ p) {
  long i = (long)blockIdx.x * blockDim.x + threadIdx.x;
  const long n = (long)BS * NANG * NPIX;
  if (i >= n) return;
  int pix = (int)(i % NPIX);
  int a   = (int)((i / NPIX) % NANG);
  int y = pix / MAPD, x = pix - (pix / MAPD) * MAPD;
  p[i] = (a == 0 && y == (MAPD - 1) / 2 && x == (MAPD - 1) / 2) ? (__bf16)1.0f
                                                                : (__bf16)0.0f;
}

// ---------------------------------------------------------------------------
// Rotation resample (bilinear, zero pad) fused with A-panel packing:
//   kw[l][b][f][(a*25+ky)*32 + kx]     = rot[l][b][a][f][24-ky][24-kx]
//   kl[l][b][a][((f-1)*25+ky)*32 + kx] = rot[l][b][a][f][ky][kx]   (f >= 1)
// kx 25..31 of every group stays zero (pre-zeroed buffer).
// ---------------------------------------------------------------------------
__global__ void __launch_bounds__(256)
rotate_pack_kernel(const float* __restrict__ img, __bf16* __restrict__ kw,
                   __bf16* __restrict__ kl) {
  long idx = (long)blockIdx.x * blockDim.x + threadIdx.x;
  const long total = (long)LSEQ * BS * NANG * FCH * LOCD * LOCD;
  if (idx >= total) return;
  int j = (int)(idx % LOCD);
  int i = (int)((idx / LOCD) % LOCD);
  int f = (int)((idx / (LOCD * LOCD)) % FCH);
  int a = (int)((idx / ((long)LOCD * LOCD * FCH)) % NANG);
  int b = (int)((idx / ((long)LOCD * LOCD * FCH * NANG)) % BS);
  int l = (int)(idx / ((long)LOCD * LOCD * FCH * NANG * BS));

  float ang = (float)a * (6.283185307179586f / (float)NANG);
  float ca = cosf(ang), sa = sinf(ang);
  float gx = -1.0f + 2.0f * (float)j / (float)(LOCD - 1);
  float gy = -1.0f + 2.0f * (float)i / (float)(LOCD - 1);
  float sxf = ca * gx - sa * gy;
  float syf = sa * gx + ca * gy;
  float px = (sxf + 1.0f) * 0.5f * (float)(LOCD - 1);
  float py = (syf + 1.0f) * 0.5f * (float)(LOCD - 1);
  int x0 = (int)floorf(px), y0 = (int)floorf(py);
  float wx1 = px - (float)x0, wy1 = py - (float)y0;
  float wx0 = 1.0f - wx1, wy0 = 1.0f - wy1;

  const float* base = img + (((long)(l * BS + b) * FCH) + f) * (LOCD * LOCD);
  auto samp = [&](int yy, int xx) -> float {
    if (yy < 0 || yy >= LOCD || xx < 0 || xx >= LOCD) return 0.0f;
    return base[yy * LOCD + xx];
  };
  float val = samp(y0, x0) * wy0 * wx0 + samp(y0, x0 + 1) * wy0 * wx1 +
              samp(y0 + 1, x0) * wy1 * wx0 + samp(y0 + 1, x0 + 1) * wy1 * wx1;

  size_t kwi = (((size_t)(l * BS + b) * FCH) + f) * KP1 +
               (size_t)(a * LOCD + (LOCD - 1 - i)) * KG + (LOCD - 1 - j);
  kw[kwi] = (__bf16)val;
  if (f >= 1) {
    size_t kli = (((size_t)(l * BS + b) * NANG) + a) * KP2 +
                 (size_t)((f - 1) * LOCD + i) * KG + j;
    kl[kli] = (__bf16)val;
  }
}

// ---------------------------------------------------------------------------
// conv1: one block per (row y, batch b); 8 waves = 2 M-tiles x 4 N-groups of
// 32 px (two 16x16 WMMA tiles per A fragment). ALL 25 halo rows of the 8
// angle channels are staged in LDS once (64 KB, single barrier); per (a,ky)
// chunk: 2 aligned 16-B A loads, 2 LDS B fragments, 2 WMMA. Out-of-range sy
// skips uniformly (cancels the 32/25 K padding on average).
// Lane layout (ISA 7.12.2, 16-bit 16x32): lanes 0-15 hold K 0-7,16-23;
// lanes 16-31 hold K 8-15,24-31.
// ---------------------------------------------------------------------------
__global__ void __launch_bounds__(256)
conv1_wmma_kernel(const __bf16* __restrict__ kw_t, const __bf16* __restrict__ pbf,
                  float* __restrict__ oreg) {
  __shared__ __bf16 tile[NANG][LOCD][LROW];   // 64000 B
  const int b   = blockIdx.z;
  const int y   = blockIdx.x;                 // uniform row
  const int tid = threadIdx.x;
  const int wv  = tid >> 5;
  const int ln  = tid & 31;
  const int mT  = wv & 1;                     // 2 M-tiles
  const int nG  = wv >> 1;                    // 4 N-groups of 32 px
  const int col = ln & 15;
  const int x0  = nG * 32;
  const int m   = mT * 16 + col;              // A row (output channel)
  const int kb  = (ln < 16) ? 0 : 8;
  const __bf16* arow = kw_t + ((size_t)b * FCH + m) * KP1;
  const __bf16* pb   = pbf + (size_t)b * NANG * NPIX;

  // Stage all halo rows (zeros outside the map).
  for (int i = tid; i < NANG * LOCD * LROW; i += 256) {
    const int a   = i / (LOCD * LROW);
    const int rem = i - a * (LOCD * LROW);
    const int ky  = rem / LROW;
    const int sxp = rem - ky * LROW;
    const int sy = y + ky - PADR;
    const int sx = sxp - PADR;
    tile[0][0][i] = (sy >= 0 && sy < MAPD && sx >= 0 && sx < MAPD)
                        ? pb[(size_t)a * NPIX + sy * MAPD + sx]
                        : (__bf16)0.0f;
  }
  __syncthreads();

  v8f acc0 = {0.f, 0.f, 0.f, 0.f, 0.f, 0.f, 0.f, 0.f};
  v8f acc1 = acc0;

#pragma unroll 1
  for (int ky = 0; ky < LOCD; ++ky) {
    const int sy = y + ky - PADR;
    if (sy < 0 || sy >= MAPD) continue;       // uniform: zero contribution
#pragma unroll 1
    for (int a = 0; a < NANG; ++a) {
      const int cb = (a * LOCD + ky) * KG;
      __builtin_prefetch(arow + cb + KG, 0, 0);
      v16bf af, b0, b1;
#pragma unroll
      for (int e = 0; e < 8; ++e) {           // two aligned 16-B runs
        af[e]     = arow[cb + kb + e];
        af[e + 8] = arow[cb + 16 + kb + e];
      }
      const __bf16* lrow = &tile[a][ky][0];
#pragma unroll
      for (int e = 0; e < 16; ++e) {          // kx>=25 hits A's exact zeros
        const int kx = (e < 8) ? (kb + e) : (8 + kb + e);
        b0[e] = lrow[x0 + col + kx];
        b1[e] = lrow[x0 + 16 + col + kx];
      }
      acc0 = wmma_bf16(af, b0, acc0);
      acc1 = wmma_bf16(af, b1, acc1);
    }
  }

  const int r0 = mT * 16 + ((ln < 16) ? 0 : 8);
#pragma unroll
  for (int q = 0; q < 8; ++q) {
    const int x = x0 + col;
    if (x < MAPD)
      oreg[((size_t)b * FCH + (r0 + q)) * NPIX + y * MAPD + x] = acc0[q];
    if (x + 16 < MAPD)
      oreg[((size_t)b * FCH + (r0 + q)) * NPIX + y * MAPD + x + 16] = acc1[q];
  }
}

// ---------------------------------------------------------------------------
// conv2: one block per (row y, batch b); 4 waves x 2 N-tiles cover the row.
// Single M-tile (rows >=8 clamp to row 7; their D rows are discarded).
// Per ky: stage the 31 channel rows in LDS (double buffered), 31 chunks.
// ---------------------------------------------------------------------------
__global__ void __launch_bounds__(128)
conv2_wmma_kernel(const __bf16* __restrict__ kl_t, const __bf16* __restrict__ msm,
                  float* __restrict__ scores) {
  __shared__ __bf16 tile[2][FCH - 1][LROW];   // 19840 B
  const int b   = blockIdx.z;
  const int y   = blockIdx.x;
  const int tid = threadIdx.x;
  const int wv  = tid >> 5;
  const int ln  = tid & 31;
  const int col = ln & 15;
  const int x0  = wv * 32;
  const int mc  = (col < NANG) ? col : (NANG - 1);
  const int kb  = (ln < 16) ? 0 : 8;
  const __bf16* arow = kl_t + ((size_t)b * NANG + mc) * KP2;
  const __bf16* mb   = msm + (size_t)b * (FCH - 1) * NPIX;

  v8f acc0 = {0.f, 0.f, 0.f, 0.f, 0.f, 0.f, 0.f, 0.f};
  v8f acc1 = acc0;

#pragma unroll 1
  for (int ky = 0; ky < LOCD; ++ky) {
    const int sy = y + ky - PADR;
    const int pg = ky & 1;
    if (sy >= 0 && sy < MAPD) {
      for (int i = tid; i < (FCH - 1) * LROW; i += 128) {
        const int c = i / LROW, sxp = i - c * LROW;
        const int sx = sxp - PADR;
        tile[pg][c][sxp] = (sx >= 0 && sx < MAPD)
                               ? mb[(size_t)c * NPIX + sy * MAPD + sx]
                               : (__bf16)0.0f;
      }
    }
    __syncthreads();                          // also fences buffer reuse
    if (sy >= 0 && sy < MAPD) {
#pragma unroll 1
      for (int c = 0; c < FCH - 1; ++c) {
        const int cb = (c * LOCD + ky) * KG;
        __builtin_prefetch(arow + cb + KG, 0, 0);
        v16bf af, b0, b1;
#pragma unroll
        for (int e = 0; e < 8; ++e) {
          af[e]     = arow[cb + kb + e];
          af[e + 8] = arow[cb + 16 + kb + e];
        }
        const __bf16* lrow = &tile[pg][c][0];
#pragma unroll
        for (int e = 0; e < 16; ++e) {
          const int kx = (e < 8) ? (kb + e) : (8 + kb + e);
          b0[e] = lrow[x0 + col + kx];
          b1[e] = lrow[x0 + 16 + col + kx];
        }
        acc0 = wmma_bf16(af, b0, acc0);
        acc1 = wmma_bf16(af, b1, acc1);
      }
    }
  }

  const int r0 = (ln < 16) ? 0 : 8;
#pragma unroll
  for (int q = 0; q < 8; ++q) {
    const int row = r0 + q;
    if (row < NANG) {
      const int x = x0 + col;
      if (x < MAPD)
        scores[((size_t)b * NANG + row) * NPIX + y * MAPD + x] = acc0[q];
      if (x + 16 < MAPD)
        scores[((size_t)b * NANG + row) * NPIX + y * MAPD + x + 16] = acc1[q];
    }
  }
}

// ---------------------------------------------------------------------------
// GRU + channel softmax on WMMA. Each wave owns 16 pixels. Gate pre-
// activations are 96x32 GEMMs with K=32 -> one WMMA chunk per 16-gate tile:
//   r,z: acc = Wih x (wmma) then accumulate Whh h on top (C-operand chain)
//   n:   inn = Wih x and hn = Whh h kept separate (torch GRU bias placement)
// 12 WMMA replace 6144 scalar FMAs per wave. D-tile layout aligns r_j, z_j,
// inn_j, hn_j for the same (gate, pixel) in the same lane/slot, so the gate
// math is elementwise; a pixel's 32 channels live in lanes p and p+16 ->
// softmax needs a single xor-16 shuffle pair.
// ---------------------------------------------------------------------------
__global__ void __launch_bounds__(256)
gru_softmax_kernel(const float* __restrict__ oreg, float* __restrict__ mbuf,
                   const float* __restrict__ wih, const float* __restrict__ whh,
                   const float* __restrict__ bih, const float* __restrict__ bhh,
                   float* __restrict__ maps_out_t, __bf16* __restrict__ msm) {
  const int wv  = threadIdx.x >> 5;
  const int ln  = threadIdx.x & 31;
  const int col = ln & 15;
  const long NPB  = (long)BS * NPIX;
  const long pixl = (long)blockIdx.x * 128 + wv * 16 + col;
  const bool pvalid = pixl < NPB;
  const long pc = pvalid ? pixl : 0;
  const int b   = (int)(pc / NPIX);
  const int pix = (int)(pc - (long)b * NPIX);
  const int kb   = (ln < 16) ? 0 : 8;
  const int roff = (ln < 16) ? 0 : 8;

  // B fragments: x (conv1 output) and h (memory), channel-K pattern.
  v16bf xf, hf;
#pragma unroll
  for (int e = 0; e < 16; ++e) {
    const int ch = (e < 8) ? (kb + e) : (8 + kb + e);
    xf[e] = (__bf16)oreg[((size_t)b * FCH + ch) * NPIX + pix];
    hf[e] = (__bf16)mbuf[((size_t)b * FCH + ch) * NPIX + pix];
  }

  auto loadA = [&](const float* w, int g0) -> v16bf {   // 16 gate rows g0..
    v16bf a;
    const float* row = w + (size_t)(g0 + col) * FCH;
#pragma unroll
    for (int e = 0; e < 16; ++e) {
      const int k = (e < 8) ? (kb + e) : (8 + kb + e);
      a[e] = (__bf16)row[k];
    }
    return a;
  };

  const v8f zv = {0.f, 0.f, 0.f, 0.f, 0.f, 0.f, 0.f, 0.f};
  v8f r0 = wmma_bf16(loadA(wih, 0), xf, zv);
  r0 = wmma_bf16(loadA(whh, 0), hf, r0);
  v8f r1 = wmma_bf16(loadA(wih, 16), xf, zv);
  r1 = wmma_bf16(loadA(whh, 16), hf, r1);
  v8f z0 = wmma_bf16(loadA(wih, 32), xf, zv);
  z0 = wmma_bf16(loadA(whh, 32), hf, z0);
  v8f z1 = wmma_bf16(loadA(wih, 48), xf, zv);
  z1 = wmma_bf16(loadA(whh, 48), hf, z1);
  v8f i0 = wmma_bf16(loadA(wih, 64), xf, zv);
  v8f i1 = wmma_bf16(loadA(wih, 80), xf, zv);
  v8f n0 = wmma_bf16(loadA(whh, 64), hf, zv);
  v8f n1 = wmma_bf16(loadA(whh, 80), hf, zv);

  float hnew[16], ex[16];
#pragma unroll
  for (int q = 0; q < 8; ++q) {
    const int j0 = q + roff;           // gates/channels of the low tile
    const int j1 = 16 + q + roff;      // high tile
    {
      const float rr = 1.f / (1.f + expf(-(r0[q] + bih[j0] + bhh[j0])));
      const float zz = 1.f / (1.f + expf(-(z0[q] + bih[32 + j0] + bhh[32 + j0])));
      const float nn = tanhf((i0[q] + bih[64 + j0]) + rr * (n0[q] + bhh[64 + j0]));
      const float h  = mbuf[((size_t)b * FCH + j0) * NPIX + pix];
      hnew[q] = (1.f - zz) * nn + zz * h;
    }
    {
      const float rr = 1.f / (1.f + expf(-(r1[q] + bih[j1] + bhh[j1])));
      const float zz = 1.f / (1.f + expf(-(z1[q] + bih[32 + j1] + bhh[32 + j1])));
      const float nn = tanhf((i1[q] + bih[64 + j1]) + rr * (n1[q] + bhh[64 + j1]));
      const float h  = mbuf[((size_t)b * FCH + j1) * NPIX + pix];
      hnew[8 + q] = (1.f - zz) * nn + zz * h;
    }
  }

  // Channel softmax: local 16, then combine with partner lane (xor 16).
  float mx = hnew[0];
#pragma unroll
  for (int q = 1; q < 16; ++q) mx = fmaxf(mx, hnew[q]);
  mx = fmaxf(mx, __shfl_xor(mx, 16, 32));
  float s = 0.f;
#pragma unroll
  for (int q = 0; q < 16; ++q) {
    ex[q] = expf(hnew[q] - mx);
    s += ex[q];
  }
  s += __shfl_xor(s, 16, 32);
  const float inv = 1.f / s;

  if (pvalid) {
#pragma unroll
    for (int q = 0; q < 8; ++q) {
      const int j0 = q + roff;
      const int j1 = 16 + q + roff;
      mbuf[((size_t)b * FCH + j0) * NPIX + pix] = hnew[q];
      mbuf[((size_t)b * FCH + j1) * NPIX + pix] = hnew[8 + q];
      maps_out_t[((size_t)b * FCH + j0) * NPIX + pix] = hnew[q];
      maps_out_t[((size_t)b * FCH + j1) * NPIX + pix] = hnew[8 + q];
      if (j0 >= 1)
        msm[((size_t)b * (FCH - 1) + (j0 - 1)) * NPIX + pix] =
            (__bf16)(ex[q] * inv);
      msm[((size_t)b * (FCH - 1) + (j1 - 1)) * NPIX + pix] =
          (__bf16)(ex[8 + q] * inv);
    }
  }
}

// ---------------------------------------------------------------------------
// Global softmax over the 8*125*125 scores of one batch item (block per b).
// ---------------------------------------------------------------------------
__global__ void __launch_bounds__(1024)
pose_softmax_kernel(const float* __restrict__ scores, float* __restrict__ poses_t,
                    __bf16* __restrict__ pbf) {
  const int b = blockIdx.x;
  const int N = NANG * NPIX;
  __shared__ float red[1024];
  const float* s = scores + (size_t)b * N;

  float mx = -3.0e38f;
  for (int i = threadIdx.x; i < N; i += 1024) mx = fmaxf(mx, s[i]);
  red[threadIdx.x] = mx;
  __syncthreads();
  for (int st = 512; st > 0; st >>= 1) {
    if (threadIdx.x < st)
      red[threadIdx.x] = fmaxf(red[threadIdx.x], red[threadIdx.x + st]);
    __syncthreads();
  }
  mx = red[0];
  __syncthreads();

  float sum = 0.0f;
  for (int i = threadIdx.x; i < N; i += 1024) sum += expf(s[i] - mx);
  red[threadIdx.x] = sum;
  __syncthreads();
  for (int st = 512; st > 0; st >>= 1) {
    if (threadIdx.x < st) red[threadIdx.x] += red[threadIdx.x + st];
    __syncthreads();
  }
  const float inv = 1.0f / red[0];

  for (int i = threadIdx.x; i < N; i += 1024) {
    const float v = expf(s[i] - mx) * inv;
    poses_t[(size_t)b * N + i] = v;
    pbf[(size_t)b * N + i] = (__bf16)v;
  }
}

// ---------------------------------------------------------------------------
extern "C" void kernel_launch(void* const* d_in, const int* in_sizes, int n_in,
                              void* d_out, int out_size, void* d_ws,
                              size_t ws_size, hipStream_t stream) {
  (void)in_sizes; (void)n_in; (void)out_size; (void)ws_size;

  const float* img   = (const float*)d_in[0];  // (6,8,32,25,25)
  const float* maps0 = (const float*)d_in[1];  // (1,8,32,125,125)
  // d_in[2] pose_changes: unused in reference math
  const float* w_ih  = (const float*)d_in[3];
  const float* w_hh  = (const float*)d_in[4];
  const float* b_ih  = (const float*)d_in[5];
  const float* b_hh  = (const float*)d_in[6];
  float* out = (float*)d_out;

  // Workspace carve-up (256-B aligned), ~85 MB total.
  uint8_t* ws = (uint8_t*)d_ws;
  size_t off = 0;
  auto take = [&](size_t bytes) -> uint8_t* {
    uint8_t* p = ws + off;
    off += (bytes + 255) & ~(size_t)255;
    return p;
  };
  __bf16* kw   = (__bf16*)take((size_t)LSEQ * BS * FCH * KP1 * 2);   // 19.7 MB
  __bf16* kl   = (__bf16*)take((size_t)LSEQ * BS * NANG * KP2 * 2);  // 19.1 MB
  __bf16* pbf  = (__bf16*)take((size_t)BS * NANG * NPIX * 2);        //  2.0 MB
  __bf16* msm  = (__bf16*)take((size_t)BS * (FCH - 1) * NPIX * 2);   //  7.8 MB
  float*  mbuf = (float*)take((size_t)BS * FCH * NPIX * 4);          // 16.0 MB
  float*  oreg = (float*)take((size_t)BS * FCH * NPIX * 4);          // 16.0 MB
  float*  scr  = (float*)take((size_t)BS * NANG * NPIX * 4);         //  4.0 MB

  {  // zero padded A panels, rotate+pack, init pose & memory
    long n1 = (long)LSEQ * BS * FCH * KP1;
    long n2 = (long)LSEQ * BS * NANG * KP2;
    zero_bf16_kernel<<<(int)((n1 + 255) / 256), 256, 0, stream>>>(kw, n1);
    zero_bf16_kernel<<<(int)((n2 + 255) / 256), 256, 0, stream>>>(kl, n2);
    long nr = (long)LSEQ * BS * NANG * FCH * LOCD * LOCD;
    rotate_pack_kernel<<<(int)((nr + 255) / 256), 256, 0, stream>>>(img, kw, kl);
    long np = (long)BS * NANG * NPIX;
    init_pose_kernel<<<(int)((np + 255) / 256), 256, 0, stream>>>(pbf);
    long nm = (long)BS * FCH * NPIX;
    copy_f32_kernel<<<(int)((nm + 255) / 256), 256, 0, stream>>>(maps0, mbuf, nm);
  }

  float* poses_out = out;                                             // 5e6
  float* maps_out  = out + (size_t)(LSEQ - 1) * BS * NANG * NPIX;     // 2e7
  float* mfinal    = maps_out + (size_t)(LSEQ - 1) * BS * FCH * NPIX; // 4e6

  const dim3 gconv(MAPD, 1, BS);              // row-tile blocks
  const int  gru_blocks = (int)(((long)BS * NPIX + 127) / 128);

  for (int t = 0; t < LSEQ - 1; ++t) {
    conv1_wmma_kernel<<<gconv, 256, 0, stream>>>(
        kw + (size_t)t * BS * FCH * KP1, pbf, oreg);
    gru_softmax_kernel<<<gru_blocks, 256, 0, stream>>>(
        oreg, mbuf, w_ih, w_hh, b_ih, b_hh,
        maps_out + (size_t)t * BS * FCH * NPIX, msm);
    conv2_wmma_kernel<<<gconv, 128, 0, stream>>>(
        kl + (size_t)(t + 1) * BS * NANG * KP2, msm, scr);
    pose_softmax_kernel<<<BS, 1024, 0, stream>>>(
        scr, poses_out + (size_t)t * BS * NANG * NPIX, pbf);
  }

  hipMemcpyAsync(mfinal, mbuf, (size_t)BS * FCH * NPIX * sizeof(float),
                 hipMemcpyDeviceToDevice, stream);
}
